// MultiHeadAttention_50053548868010
// MI455X (gfx1250) — matrix operand order repaired and checked by reading, not compiled, not run on hardware
//
#include <hip/hip_runtime.h>

// ---------------------------------------------------------------------------
// MHA block for MI455X (gfx1250, wave32, WMMA).
// B=4, L=2048, DX=1024, H=16, DK=DV=64.
// GEMMs: v_wmma_f32_16x16x32_bf16 (f32 accumulate).
// Projection/output GEMMs stage tiles with GLOBAL_LOAD_ASYNC_TO_LDS_B128
// (ASYNCcnt-tracked DMA) and double-buffered LDS.
// ---------------------------------------------------------------------------

#define B_   4
#define L_   2048
#define DX_  1024
#define H_   16
#define DKV_ 64
#define M_   (B_ * L_)          // 8192 rows of the [M, 1024] activations
#define MK_  (M_ * DX_)         // 8,388,608 elements

typedef __attribute__((ext_vector_type(16))) __bf16 bf16x16;
typedef __attribute__((ext_vector_type(8)))  __bf16 bf16x8;
typedef __attribute__((ext_vector_type(8)))  float  floatx8;

union FragAB {
  bf16x16 v;
  bf16x8  h[2];
  uint4   q[2];
};

__device__ __forceinline__ unsigned short f2bf(float f) {
  unsigned int u = __float_as_uint(f);
  u += 0x7FFFu + ((u >> 16) & 1u);          // round-to-nearest-even
  return (unsigned short)(u >> 16);
}
__device__ __forceinline__ unsigned int pk2(float lo, float hi) {
  return (unsigned int)f2bf(lo) | ((unsigned int)f2bf(hi) << 16);
}
__device__ __forceinline__ floatx8 zero8() {
  floatx8 z = {0.f, 0.f, 0.f, 0.f, 0.f, 0.f, 0.f, 0.f};
  return z;
}
__device__ __forceinline__ floatx8 wmma_bf16(const FragAB& a, const FragAB& b, floatx8 c) {
  // (neg_a, A, neg_b, B, c_mod, C, reuse_a, reuse_b)
  return __builtin_amdgcn_wmma_f32_16x16x32_bf16(false, a.v, false, b.v,
                                                 (short)0, c, false, false);
}

// Async DMA: 16 bytes global -> LDS per lane, tracked by ASYNCcnt.
// LDS address operand = low 32 bits of the flat shared pointer
// (ISA 10.2).
__device__ __forceinline__ void async_b128(unsigned short* lds_dst,
                                           const unsigned short* g) {
  unsigned loff = (unsigned)(unsigned long long)lds_dst;
  asm volatile("global_load_async_to_lds_b128 %0, %1, off"
               :: "v"(loff), "v"(g) : "memory");
}

// ---------------------------------------------------------------------------
// fp32 -> bf16 elementwise (n divisible by 1024)
// ---------------------------------------------------------------------------
__global__ void __launch_bounds__(256) cvt_f32_bf16(const float* __restrict__ src,
                                                    unsigned short* __restrict__ dst,
                                                    int n) {
  int i = (blockIdx.x * 256 + threadIdx.x) * 4;
  if (i + 3 < n) {
    float4 f = *(const float4*)(src + i);
    uint2 o;
    o.x = pk2(f.x, f.y);
    o.y = pk2(f.z, f.w);
    *(uint2*)(dst + i) = o;
  }
}

// Transpose-convert weight: src [1024][1024] fp32 (input-major, y=x@W),
// dst [n][k] bf16 so GEMM B-operand is K-contiguous per output column.
__global__ void __launch_bounds__(256) cvt_wT(const float* __restrict__ src,
                                              unsigned short* __restrict__ dst) {
  int idx = blockIdx.x * 256 + threadIdx.x;   // k fastest -> coalesced writes
  int k = idx & 1023, n = idx >> 10;
  dst[idx] = f2bf(src[k * 1024 + n]);
}

// ---------------------------------------------------------------------------
// Generic bf16 WMMA GEMM: C[M,1024] = A[M,1024] @ Bt^T + bias
//   A  : bf16 row-major [M, 1024]
//   Bt : bf16 [1024, 1024], row n holds column n of W (K-contiguous)
// Block tile 128x128, 8 waves in 4(M)x2(N), each wave 32x64 = 2x4 WMMA tiles.
// Global->LDS staging via async DMA, double-buffered; per K-step each wave
// issues 4 async b128 (A:2, B:2); s_wait_asynccnt 4 retires tile k while the
// prefetch of tile k+1 stays in flight.
// mode 0: bf16 natural out, scaled by alpha (q/k projections)
// mode 1: bf16 head-transposed out [B,H,DV,L]   (v projection)
// mode 2: fp32 out = acc + bias + residual       (output projection)
// ---------------------------------------------------------------------------
#define KP 40   // LDS row pitch (ushorts): 80B -> conflict-free b128 frags

__global__ void __launch_bounds__(256) gemm_bf16(
    const unsigned short* __restrict__ A,
    const unsigned short* __restrict__ Bt,
    const float* __restrict__ bias,
    const float* __restrict__ residual,
    void* __restrict__ Cout,
    float alpha, int mode) {
  __shared__ unsigned short As[2][128 * KP];
  __shared__ unsigned short Bs[2][128 * KP];

  const int tid = threadIdx.x;
  const int lane = tid & 31, wid = tid >> 5;
  const int wm = wid & 3, wn = wid >> 2;         // 4 x 2 wave grid
  const int l15 = lane & 15, hi = (lane >> 4) & 1;
  const int m0 = blockIdx.y * 128, n0 = blockIdx.x * 128;

  const int rF = tid >> 1, cF = (tid & 1) * 16;  // tile-fill coords (128x32)
  const unsigned short* gA = A  + (size_t)(m0 + rF) * 1024 + cF;
  const unsigned short* gB = Bt + (size_t)(n0 + rF) * 1024 + cF;

  floatx8 acc[2][4];
#pragma unroll
  for (int i = 0; i < 2; ++i)
#pragma unroll
    for (int j = 0; j < 4; ++j) acc[i][j] = zero8();

  // Prologue: prefetch tile 0 into buffer 0.
  {
    async_b128(&As[0][rF * KP + cF],     gA);
    async_b128(&As[0][rF * KP + cF + 8], gA + 8);
    async_b128(&Bs[0][rF * KP + cF],     gB);
    async_b128(&Bs[0][rF * KP + cF + 8], gB + 8);
  }

  for (int kt = 0; kt < 1024; kt += 32) {
    const int cur = (kt >> 5) & 1;
    // Everyone done reading the buffer we are about to overwrite.
    __syncthreads();
    if (kt + 32 < 1024) {
      const int nxt = cur ^ 1;
      async_b128(&As[nxt][rF * KP + cF],     gA + kt + 32);
      async_b128(&As[nxt][rF * KP + cF + 8], gA + kt + 40);
      async_b128(&Bs[nxt][rF * KP + cF],     gB + kt + 32);
      async_b128(&Bs[nxt][rF * KP + cF + 8], gB + kt + 40);
      asm volatile("s_wait_asynccnt 0x4" ::: "memory");  // tile k done
    } else {
      asm volatile("s_wait_asynccnt 0x0" ::: "memory");
    }
    __syncthreads();   // all waves' DMA for tile k visible in LDS

    FragAB af[2], bfr[4];
#pragma unroll
    for (int i = 0; i < 2; ++i) {
      int ar = wm * 32 + i * 16 + l15;
      int ac = hi * 16;                           // A: K contiguous 16
      af[i].h[0] = *(const bf16x8*)&As[cur][ar * KP + ac];
      af[i].h[1] = *(const bf16x8*)&As[cur][ar * KP + ac + 8];
    }
#pragma unroll
    for (int j = 0; j < 4; ++j) {
      int br = wn * 64 + j * 16 + l15;
      int bc = hi * 16;                           // B: K contiguous 16
      bfr[j].h[0] = *(const bf16x8*)&Bs[cur][br * KP + bc];
      bfr[j].h[1] = *(const bf16x8*)&Bs[cur][br * KP + bc + 8];
    }
#pragma unroll
    for (int i = 0; i < 2; ++i)
#pragma unroll
      for (int j = 0; j < 4; ++j)
        acc[i][j] = wmma_bf16(af[i], bfr[j], acc[i][j]);
  }

#pragma unroll
  for (int i = 0; i < 2; ++i)
#pragma unroll
    for (int j = 0; j < 4; ++j)
#pragma unroll
      for (int v = 0; v < 8; ++v) {
        int m = m0 + wm * 32 + i * 16 + hi * 8 + v;
        int n = n0 + wn * 64 + j * 16 + l15;
        float r = acc[i][j][v] + bias[n];
        if (mode == 0) {
          ((unsigned short*)Cout)[(size_t)m * 1024 + n] = f2bf(r * alpha);
        } else if (mode == 1) {
          int b = m >> 11, l = m & (L_ - 1);
          int h = n >> 6, dv = n & 63;
          ((unsigned short*)Cout)[((size_t)(b * H_ + h) * 64 + dv) * L_ + l] = f2bf(r);
        } else {
          ((float*)Cout)[(size_t)m * 1024 + n] = r + residual[(size_t)m * 1024 + n];
        }
      }
}

// ---------------------------------------------------------------------------
// Fused scores + causal softmax. Block = (b, h, 16 query rows).
// q is pre-scaled by 1/sqrt(DK). Scores kept in 16x2052 fp32 LDS (131 KB of
// the 320 KB WGP LDS), then row softmax and full 2048-wide coalesced write.
// B fragments feed WMMA straight from global (contiguous 32B per lane).
// ---------------------------------------------------------------------------
#define SP 2052

__global__ void __launch_bounds__(256) attn_scores_softmax(
    const unsigned short* __restrict__ qb,   // [B*L, 1024] bf16 (scaled)
    const unsigned short* __restrict__ kb,   // [B*L, 1024] bf16
    float* __restrict__ att) {               // [H*B, L, L] fp32
  extern __shared__ float sc[];              // [16][SP]
  const int tid = threadIdx.x;
  const int lane = tid & 31, wid = tid >> 5;
  const int l15 = lane & 15, hi = (lane >> 4) & 1;
  const int q0 = blockIdx.x * 16;
  const int bh = blockIdx.y;
  const int b = bh >> 4, h = bh & (H_ - 1);

  // A fragments: 16 q-rows x K=64 -> two k-steps, loaded straight from global
  FragAB af[2];
  {
    const unsigned short* ga =
        qb + (size_t)(b * L_ + q0 + l15) * 1024 + h * DKV_;
    int ac = hi * 16;
    af[0].h[0] = *(const bf16x8*)(ga + ac);
    af[0].h[1] = *(const bf16x8*)(ga + ac + 8);
    af[1].h[0] = *(const bf16x8*)(ga + 32 + ac);
    af[1].h[1] = *(const bf16x8*)(ga + 32 + ac + 8);
  }

  const int nchunks = (q0 + 16 + 127) >> 7;   // causal: keys <= q0+15
  for (int c = 0; c < nchunks; ++c) {
    const int key = (c << 7) + wid * 16 + l15;
    const unsigned short* gb =
        kb + (size_t)(b * L_ + key) * 1024 + h * DKV_ + hi * 16;
    FragAB b0, b1;                             // contiguous 32B per lane
    b0.q[0] = *(const uint4*)gb;
    b0.q[1] = *(const uint4*)(gb + 8);
    b1.q[0] = *(const uint4*)(gb + 32);
    b1.q[1] = *(const uint4*)(gb + 40);
    floatx8 acc = zero8();
    acc = wmma_bf16(af[0], b0, acc);
    acc = wmma_bf16(af[1], b1, acc);
#pragma unroll
    for (int v = 0; v < 8; ++v) {
      int r = hi * 8 + v;                      // query row within tile
      sc[r * SP + key] = (key <= q0 + r) ? acc[v] : -3.0e38f;
    }
  }
  __syncthreads();

  // Softmax: wave w owns rows 2w, 2w+1
#pragma unroll
  for (int rr = 0; rr < 2; ++rr) {
    const int r = wid * 2 + rr, q = q0 + r;
    float* row = sc + r * SP;
    float mx = -3.0e38f;
    for (int c2 = lane; c2 <= q; c2 += 32) mx = fmaxf(mx, row[c2]);
#pragma unroll
    for (int off = 16; off; off >>= 1) mx = fmaxf(mx, __shfl_xor(mx, off, 32));
    float sum = 0.f;
    for (int c2 = lane; c2 <= q; c2 += 32) {
      float e = __expf(row[c2] - mx);
      row[c2] = e;
      sum += e;
    }
#pragma unroll
    for (int off = 16; off; off >>= 1) sum += __shfl_xor(sum, off, 32);
    const float inv = 1.f / sum;
    float* g = att + ((size_t)(h * B_ + b) * L_ + q) * L_;
    for (int c2 = lane; c2 < L_; c2 += 32)
      g[c2] = (c2 <= q) ? row[c2] * inv : 0.f;   // zeros past the diagonal
  }
}

// ---------------------------------------------------------------------------
// ctx = att @ v. A = fp32 att (converted to bf16 into LDS), B = vpt [64, L].
// Block tile 128x64, wave w owns rows w*16..w*16+15 (1x4 WMMA tiles).
// Causality: K-loop stops at m0+128 (att is zero beyond the diagonal).
// ---------------------------------------------------------------------------
__global__ void __launch_bounds__(256) attn_av(
    const float* __restrict__ att,            // [H*B, L, L]
    const unsigned short* __restrict__ vpt,   // [B*H, 64, L] bf16
    unsigned short* __restrict__ ctx) {       // [B*L, 1024] bf16
  __shared__ unsigned short As[128 * KP];
  __shared__ unsigned short Bs[64 * KP];
  const int tid = threadIdx.x;
  const int lane = tid & 31, wid = tid >> 5;
  const int l15 = lane & 15, hi = (lane >> 4) & 1;
  const int m0 = blockIdx.x * 128;
  const int bh = blockIdx.y;
  const int b = bh >> 4, h = bh & (H_ - 1);
  const float* Ab = att + (size_t)(h * B_ + b) * L_ * L_;
  const unsigned short* Bb = vpt + (size_t)bh * 64 * L_;

  floatx8 acc[4];
#pragma unroll
  for (int j = 0; j < 4; ++j) acc[j] = zero8();

  const int rA = tid >> 1, cA = (tid & 1) * 16;  // A fill 128x32
  const int rB = tid >> 2, cB = (tid & 3) * 8;   // B fill 64x32
  const int kmax = m0 + 128;

  for (int kt = 0; kt < kmax; kt += 32) {
    __syncthreads();
    {
      const float* ga = Ab + (size_t)(m0 + rA) * L_ + kt + cA;
      float4 f0 = *(const float4*)ga;
      float4 f1 = *(const float4*)(ga + 4);
      float4 f2 = *(const float4*)(ga + 8);
      float4 f3 = *(const float4*)(ga + 12);
      uint4 o0, o1;
      o0.x = pk2(f0.x, f0.y); o0.y = pk2(f0.z, f0.w);
      o0.z = pk2(f1.x, f1.y); o0.w = pk2(f1.z, f1.w);
      o1.x = pk2(f2.x, f2.y); o1.y = pk2(f2.z, f2.w);
      o1.z = pk2(f3.x, f3.y); o1.w = pk2(f3.z, f3.w);
      *(uint4*)&As[rA * KP + cA]     = o0;
      *(uint4*)&As[rA * KP + cA + 8] = o1;
      const unsigned short* gb = Bb + (size_t)rB * L_ + kt + cB;
      *(uint4*)&Bs[rB * KP + cB] = *(const uint4*)gb;
    }
    __syncthreads();

    FragAB a;
    {
      int ar = wid * 16 + l15;
      int ac = hi * 16;
      a.h[0] = *(const bf16x8*)&As[ar * KP + ac];
      a.h[1] = *(const bf16x8*)&As[ar * KP + ac + 8];
    }
#pragma unroll
    for (int j = 0; j < 4; ++j) {
      FragAB bb;
      int br = j * 16 + l15;
      int bc = hi * 16;
      bb.h[0] = *(const bf16x8*)&Bs[br * KP + bc];
      bb.h[1] = *(const bf16x8*)&Bs[br * KP + bc + 8];
      acc[j] = wmma_bf16(a, bb, acc[j]);
    }
  }

#pragma unroll
  for (int j = 0; j < 4; ++j)
#pragma unroll
    for (int v = 0; v < 8; ++v) {
      int l = m0 + wid * 16 + hi * 8 + v;
      int dv = j * 16 + l15;
      ctx[(size_t)(b * L_ + l) * 1024 + h * DKV_ + dv] = f2bf(acc[j][v]);
    }
}

// ---------------------------------------------------------------------------
// LayerNorm over DX=1024 per row; block = row, 256 threads x 4 elems.
// ---------------------------------------------------------------------------
__global__ void __launch_bounds__(256) ln_kernel(const float* __restrict__ x,
                                                 const float* __restrict__ gamma,
                                                 const float* __restrict__ beta,
                                                 float* __restrict__ y) {
  __shared__ float red[16];
  const int tid = threadIdx.x, lane = tid & 31, wid = tid >> 5;
  const size_t row = blockIdx.x;
  const float* xr = x + row * 1024;
  float4 v = *(const float4*)(xr + tid * 4);
  float s = v.x + v.y + v.z + v.w;
  float ss = v.x * v.x + v.y * v.y + v.z * v.z + v.w * v.w;
#pragma unroll
  for (int off = 16; off; off >>= 1) {
    s += __shfl_xor(s, off, 32);
    ss += __shfl_xor(ss, off, 32);
  }
  if (lane == 0) { red[wid] = s; red[8 + wid] = ss; }
  __syncthreads();
  float ts = 0.f, tss = 0.f;
#pragma unroll
  for (int w = 0; w < 8; ++w) { ts += red[w]; tss += red[8 + w]; }
  const float mu = ts * (1.f / 1024.f);
  const float var = tss * (1.f / 1024.f) - mu * mu;
  const float inv = rsqrtf(var + 1e-5f);
  float4 g = *(const float4*)(gamma + tid * 4);
  float4 bb = *(const float4*)(beta + tid * 4);
  float4 o;
  o.x = (v.x - mu) * inv * g.x + bb.x;
  o.y = (v.y - mu) * inv * g.y + bb.y;
  o.z = (v.z - mu) * inv * g.z + bb.z;
  o.w = (v.w - mu) * inv * g.w + bb.w;
  *(float4*)(y + row * 1024 + tid * 4) = o;
}

// ---------------------------------------------------------------------------
extern "C" void kernel_launch(void* const* d_in, const int* in_sizes, int n_in,
                              void* d_out, int out_size, void* d_ws, size_t ws_size,
                              hipStream_t stream) {
  const float* Q  = (const float*)d_in[0];
  const float* K  = (const float*)d_in[1];
  const float* V  = (const float*)d_in[2];
  // d_in[3] = causal mask, computed analytically instead
  const float* Wq = (const float*)d_in[4];
  const float* bq = (const float*)d_in[5];
  const float* Wk = (const float*)d_in[6];
  const float* bk = (const float*)d_in[7];
  const float* Wv = (const float*)d_in[8];
  const float* bv = (const float*)d_in[9];
  const float* Wo = (const float*)d_in[10];
  const float* bo = (const float*)d_in[11];
  const float* gamma = (const float*)d_in[12];
  const float* beta  = (const float*)d_in[13];

  float* y   = (float*)d_out;
  float* att = y + (size_t)M_ * DX_;           // [H*B, L, L]

  // Workspace carve-up (~152 MB)
  char* w = (char*)d_ws;
  auto carve = [&](size_t bytes) {
    char* p = w;
    w += (bytes + 255) & ~(size_t)255;
    return p;
  };
  unsigned short* Qb  = (unsigned short*)carve((size_t)MK_ * 2);
  unsigned short* Kb  = (unsigned short*)carve((size_t)MK_ * 2);
  unsigned short* Vb  = (unsigned short*)carve((size_t)MK_ * 2);
  unsigned short* Wqt = (unsigned short*)carve((size_t)DX_ * 1024 * 2);
  unsigned short* Wkt = (unsigned short*)carve((size_t)DX_ * 1024 * 2);
  unsigned short* Wvt = (unsigned short*)carve((size_t)DX_ * 1024 * 2);
  unsigned short* Wot = (unsigned short*)carve((size_t)DX_ * 1024 * 2);
  unsigned short* qp  = (unsigned short*)carve((size_t)MK_ * 2);  // [B*L,1024]
  unsigned short* kp  = (unsigned short*)carve((size_t)MK_ * 2);
  unsigned short* vpt = (unsigned short*)carve((size_t)MK_ * 2);  // [B*H,64,L]
  unsigned short* ctx = (unsigned short*)carve((size_t)MK_ * 2);
  float* xres = (float*)carve((size_t)MK_ * 4);

  const dim3 blk(256);

  cvt_f32_bf16<<<MK_ / 1024, blk, 0, stream>>>(Q, Qb, MK_);
  cvt_f32_bf16<<<MK_ / 1024, blk, 0, stream>>>(K, Kb, MK_);
  cvt_f32_bf16<<<MK_ / 1024, blk, 0, stream>>>(V, Vb, MK_);
  cvt_wT<<<4096, blk, 0, stream>>>(Wq, Wqt);
  cvt_wT<<<4096, blk, 0, stream>>>(Wk, Wkt);
  cvt_wT<<<4096, blk, 0, stream>>>(Wv, Wvt);
  cvt_wT<<<4096, blk, 0, stream>>>(Wo, Wot);

  const dim3 ggrid(1024 / 128, M_ / 128);      // (8, 64)
  gemm_bf16<<<ggrid, blk, 0, stream>>>(Qb, Wqt, bq, nullptr, qp, 0.125f, 0);
  gemm_bf16<<<ggrid, blk, 0, stream>>>(Kb, Wkt, bk, nullptr, kp, 1.0f, 0);
  gemm_bf16<<<ggrid, blk, 0, stream>>>(Vb, Wvt, bv, nullptr, vpt, 1.0f, 1);

  attn_scores_softmax<<<dim3(L_ / 16, B_ * H_), blk, 16 * SP * sizeof(float),
                        stream>>>(qp, kp, att);

  attn_av<<<dim3(L_ / 128, B_ * H_), blk, 0, stream>>>(att, vpt, ctx);

  gemm_bf16<<<ggrid, blk, 0, stream>>>(ctx, Wot, bo, Q, xres, 1.0f, 2);

  ln_kernel<<<M_, blk, 0, stream>>>(xres, gamma, beta, y);
}
